// QLinear_66752381714750
// MI455X (gfx1250) — 8-bit matrix operand order repaired and checked by reading, compiled, not run on hardware
//
#include <hip/hip_runtime.h>

typedef __attribute__((ext_vector_type(8))) int v8i;
typedef __attribute__((ext_vector_type(4))) int v4i;

#define TILE_M 128
#define TILE_N 128
#define TILE_K 128
#define N_ROWS 8192
#define IN_DIM 4096
#define OUT_DIM 4096

#define GLOBAL_AS __attribute__((address_space(1)))
#define LDS_AS    __attribute__((address_space(3)))

#if defined(__HIP_DEVICE_COMPILE__) && defined(__has_builtin)
#if __has_builtin(__builtin_amdgcn_global_load_async_to_lds_b128)
#define USE_ASYNC 1
#endif
#endif
#ifndef USE_ASYNC
#define USE_ASYNC 0
#endif

#if USE_ASYNC
#if __has_builtin(__builtin_amdgcn_s_wait_asynccnt)
#define WAIT_ASYNC(n) __builtin_amdgcn_s_wait_asynccnt(n)
#else
#define WAIT_ASYNC(n) asm volatile("s_wait_asynccnt %0" ::"i"(n) : "memory")
#endif
#else
#define WAIT_ASYNC(n)
#endif

__global__ __launch_bounds__(256) void absmax_kernel(const float* __restrict__ in,
                                                     unsigned* __restrict__ out_bits,
                                                     int n4) {
    __shared__ float red[256];
    float m = 0.0f;
    const int stride = gridDim.x * blockDim.x;
    for (int i = blockIdx.x * blockDim.x + threadIdx.x; i < n4; i += stride) {
        float4 v = ((const float4*)in)[i];
        m = fmaxf(m, fmaxf(fmaxf(fabsf(v.x), fabsf(v.y)),
                           fmaxf(fabsf(v.z), fabsf(v.w))));
    }
    red[threadIdx.x] = m;
    __syncthreads();
    for (int s = 128; s > 0; s >>= 1) {
        if ((int)threadIdx.x < s)
            red[threadIdx.x] = fmaxf(red[threadIdx.x], red[threadIdx.x + s]);
        __syncthreads();
    }
    if (threadIdx.x == 0) atomicMax(out_bits, __float_as_uint(red[0]));
}

__global__ __launch_bounds__(256) void quant_kernel(const float* __restrict__ in,
                                                    signed char* __restrict__ q,
                                                    const unsigned* __restrict__ amax_bits,
                                                    int n16) {
    int i = blockIdx.x * blockDim.x + threadIdx.x;
    if (i >= n16) return;
    const float inv = 127.0f / __uint_as_float(*amax_bits);
    const float4* p = (const float4*)in + (size_t)i * 4;
    int packed[4];
#pragma unroll
    for (int j = 0; j < 4; ++j) {
        float4 v = p[j];
        int q0 = (int)rintf(fminf(fmaxf(v.x * inv, -127.0f), 127.0f));
        int q1 = (int)rintf(fminf(fmaxf(v.y * inv, -127.0f), 127.0f));
        int q2 = (int)rintf(fminf(fmaxf(v.z * inv, -127.0f), 127.0f));
        int q3 = (int)rintf(fminf(fmaxf(v.w * inv, -127.0f), 127.0f));
        packed[j] = (q0 & 255) | ((q1 & 255) << 8) | ((q2 & 255) << 16) | ((q3 & 255) << 24);
    }
    *(int4*)(q + (size_t)i * 16) = make_int4(packed[0], packed[1], packed[2], packed[3]);
}

__device__ __forceinline__ void stage64(const signed char* __restrict__ g,
                                        signed char* l) {
#if USE_ASYNC
#pragma unroll
    for (int c = 0; c < 4; ++c)
        __builtin_amdgcn_global_load_async_to_lds_b128(
            (GLOBAL_AS v4i*)(g + c * 16), (LDS_AS v4i*)(l + c * 16), 0, 0);
#else
    int4 t0 = ((const int4*)g)[0];
    int4 t1 = ((const int4*)g)[1];
    int4 t2 = ((const int4*)g)[2];
    int4 t3 = ((const int4*)g)[3];
    ((int4*)l)[0] = t0;
    ((int4*)l)[1] = t1;
    ((int4*)l)[2] = t2;
    ((int4*)l)[3] = t3;
#endif
}

__global__ __launch_bounds__(256) void gemm_iu8_kernel(const signed char* __restrict__ qa,
                                                       const signed char* __restrict__ qw,
                                                       const float* __restrict__ bias,
                                                       const unsigned* __restrict__ amax_bits,
                                                       float* __restrict__ out) {
    __shared__ signed char sA[2][TILE_M * TILE_K];
    __shared__ signed char sB[2][TILE_N * TILE_K];

    const int tid  = threadIdx.x;
    const int lane = tid & 31;
    const int wave = tid >> 5;
    const int wm   = wave & 1;
    const int wn   = wave >> 1;
    const int blockN = blockIdx.x;
    const int blockM = blockIdx.y;

    const int ldrow = tid >> 1;
    const int ldoff = (tid & 1) * 64;
    const signed char* gA = qa + (size_t)(blockM * TILE_M + ldrow) * IN_DIM + ldoff;
    const signed char* gB = qw + (size_t)(blockN * TILE_N + ldrow) * IN_DIM + ldoff;
    const int lofs = ldrow * TILE_K + ldoff;

    const int frow  = lane & 15;
    const int abase = (lane & 16) ? 8 : 0;
    const int bbase = (lane & 16) ? 8 : 0;

    union Frag { v8i v; int2 d2[4]; int4 d4[2]; };

    v8i acc[4][2];
#pragma unroll
    for (int ms = 0; ms < 4; ++ms)
#pragma unroll
        for (int ns = 0; ns < 2; ++ns)
            acc[ms][ns] = (v8i){0, 0, 0, 0, 0, 0, 0, 0};

    const int NT = IN_DIM / TILE_K;

    stage64(gA, &sA[0][lofs]);
    stage64(gB, &sB[0][lofs]);

    for (int t = 0; t < NT; ++t) {
        const int buf = t & 1;

        if (t + 1 < NT) {
            stage64(gA + (t + 1) * TILE_K, &sA[buf ^ 1][lofs]);
            stage64(gB + (t + 1) * TILE_K, &sB[buf ^ 1][lofs]);
            WAIT_ASYNC(8);
        } else {
            WAIT_ASYNC(0);
        }
        __syncthreads();

#pragma unroll
        for (int j = 0; j < 2; ++j) {
            const int kb = j * 64;
            Frag fa[4], fb[2];
#pragma unroll
            for (int ms = 0; ms < 4; ++ms) {
                const signed char* ap =
                    &sA[buf][(wm * 64 + ms * 16 + frow) * TILE_K + kb + abase];
                fa[ms].d2[0] = *(const int2*)(ap);
                fa[ms].d2[1] = *(const int2*)(ap + 16);
                fa[ms].d2[2] = *(const int2*)(ap + 32);
                fa[ms].d2[3] = *(const int2*)(ap + 48);
            }
#pragma unroll
            for (int ns = 0; ns < 2; ++ns) {
                const signed char* bp =
                    &sB[buf][(wn * 32 + ns * 16 + frow) * TILE_K + kb + bbase];
                fb[ns].d2[0] = *(const int2*)(bp);
                fb[ns].d2[1] = *(const int2*)(bp + 16);
                fb[ns].d2[2] = *(const int2*)(bp + 32);
                fb[ns].d2[3] = *(const int2*)(bp + 48);
            }
#pragma unroll
            for (int ns = 0; ns < 2; ++ns)
#pragma unroll
                for (int ms = 0; ms < 4; ++ms)
                    acc[ms][ns] = __builtin_amdgcn_wmma_i32_16x16x64_iu8(
                        true, fa[ms].v, true, fb[ns].v, acc[ms][ns], false, false);
        }
        __syncthreads();
    }

    const float scale = (__uint_as_float(amax_bits[0]) * (1.0f / 127.0f)) *
                        (__uint_as_float(amax_bits[1]) * (1.0f / 127.0f));
    const int nlocal = lane & 15;
    const int mhalf  = (lane & 16) >> 1;
#pragma unroll
    for (int ns = 0; ns < 2; ++ns) {
        const int gn = blockN * TILE_N + wn * 32 + ns * 16 + nlocal;
        const float bv = bias[gn];
#pragma unroll
        for (int ms = 0; ms < 4; ++ms) {
            const int gm0 = blockM * TILE_M + wm * 64 + ms * 16 + mhalf;
#pragma unroll
            for (int r = 0; r < 8; ++r)
                out[(size_t)(gm0 + r) * OUT_DIM + gn] =
                    scale * (float)acc[ms][ns][r] + bv;
        }
    }
}

extern "C" void kernel_launch(void* const* d_in, const int* in_sizes, int n_in,
                              void* d_out, int out_size, void* d_ws, size_t ws_size,
                              hipStream_t stream) {
    const float* x    = (const float*)d_in[0];
    const float* w    = (const float*)d_in[1];
    const float* bias = (const float*)d_in[2];
    float* out = (float*)d_out;

    unsigned* amax = (unsigned*)d_ws;
    signed char* qa = (signed char*)d_ws + 256;
    signed char* qw = qa + (size_t)N_ROWS * IN_DIM;

    (void)hipMemsetAsync(d_ws, 0, 8, stream);

    absmax_kernel<<<2048, 256, 0, stream>>>(x, amax + 0, N_ROWS * IN_DIM / 4);
    absmax_kernel<<<1024, 256, 0, stream>>>(w, amax + 1, OUT_DIM * IN_DIM / 4);

    quant_kernel<<<(N_ROWS * IN_DIM / 16) / 256, 256, 0, stream>>>(
        x, qa, amax + 0, N_ROWS * IN_DIM / 16);
    quant_kernel<<<(OUT_DIM * IN_DIM / 16) / 256, 256, 0, stream>>>(
        w, qw, amax + 1, OUT_DIM * IN_DIM / 16);

    dim3 grid(OUT_DIM / TILE_N, N_ROWS / TILE_M);
    gemm_iu8_kernel<<<grid, 256, 0, stream>>>(qa, qw, bias, amax, out);
}
